// SingleKernelMatch_62457414419066
// MI455X (gfx1250) — matrix operand order repaired and checked by reading, not compiled, not run on hardware
//
#include <hip/hip_runtime.h>
#include <hip/hip_bf16.h>
#include <math.h>

typedef __bf16 bf16_t;
typedef __attribute__((ext_vector_type(16))) __bf16 v16bf;
typedef __attribute__((ext_vector_type(8)))  float  v8f;

#define BATCH 2
#define NPTS  4096
#define NCH   32
#define DREAL 99          // (32+1)*3
#define DPAD  128         // padded K
#define KB    4           // DPAD / 32
#define TILES (NPTS/16)   // 256 16-wide tiles per batch per side
#define INV_2ELL2 (1.0f/200.0f)   // 1/(2*ell^2), ell=10
#define THRESH 0.1f

// packed elements per side-array: BATCH * TILES * KB * 32 lanes * 16 bf16
#define PACK_ELEMS ((size_t)BATCH * TILES * KB * 32 * 16)

// ---------------------------------------------------------------------------
// Kernel 1: gather features, rotate y-side, bf16 hi/lo split, scatter into
// WMMA-ready lane layout, compute fp32 row norms.
// One thread per (side, b, n): 2*2*4096 = 16384 threads.
// ---------------------------------------------------------------------------
__global__ void prep_pack(const float* __restrict__ pc1,
                          const float* __restrict__ pc2,
                          const float* __restrict__ xfeat,
                          const float* __restrict__ yfeat,
                          const float* __restrict__ Tq,
                          bf16_t* __restrict__ Axh, bf16_t* __restrict__ Axl,
                          bf16_t* __restrict__ Byh, bf16_t* __restrict__ Byl,
                          float* __restrict__ nx,  float* __restrict__ ny)
{
    int t    = blockIdx.x * blockDim.x + threadIdx.x;   // 0..16383
    int side = t >> 13;          // 0 = x-side, 1 = y-side  (B*N = 8192)
    int bn   = t & 8191;
    int b    = bn >> 12;         // /4096
    int n    = bn & 4095;

    const float* pc = side ? pc2  : pc1;
    const float* ft = side ? yfeat : xfeat;

    // rotation matrix from pypose (x,y,z,w) quaternion (only used for y-side)
    float R[9] = {1.f,0.f,0.f, 0.f,1.f,0.f, 0.f,0.f,1.f};
    if (side) {
        float qx = Tq[b*4+0], qy = Tq[b*4+1], qz = Tq[b*4+2], qw = Tq[b*4+3];
        float inv = rsqrtf(qx*qx + qy*qy + qz*qz + qw*qw);
        qx *= inv; qy *= inv; qz *= inv; qw *= inv;
        R[0] = 1.f - 2.f*(qy*qy + qz*qz); R[1] = 2.f*(qx*qy - qz*qw); R[2] = 2.f*(qx*qz + qy*qw);
        R[3] = 2.f*(qx*qy + qz*qw); R[4] = 1.f - 2.f*(qx*qx + qz*qz); R[5] = 2.f*(qy*qz - qx*qw);
        R[6] = 2.f*(qx*qz - qy*qw); R[7] = 2.f*(qy*qz + qx*qw); R[8] = 1.f - 2.f*(qx*qx + qy*qy);
    }

    float f[DPAD];
    #pragma unroll 1
    for (int c = 0; c < NCH + 1; ++c) {
        float v0, v1, v2;
        if (c == 0) {
            v0 = pc[((size_t)b*3 + 0) * NPTS + n];
            v1 = pc[((size_t)b*3 + 1) * NPTS + n];
            v2 = pc[((size_t)b*3 + 2) * NPTS + n];
        } else {
            size_t base = (((size_t)b*NCH + (c-1)) * 3) * NPTS + n;
            v0 = ft[base];
            v1 = ft[base + NPTS];
            v2 = ft[base + 2*(size_t)NPTS];
        }
        if (side) {
            float w0 = R[0]*v0 + R[1]*v1 + R[2]*v2;
            float w1 = R[3]*v0 + R[4]*v1 + R[5]*v2;
            float w2 = R[6]*v0 + R[7]*v1 + R[8]*v2;
            v0 = w0; v1 = w1; v2 = w2;
        }
        f[c*3+0] = v0; f[c*3+1] = v1; f[c*3+2] = v2;
    }
    #pragma unroll
    for (int k = DREAL; k < DPAD; ++k) f[k] = 0.f;

    float norm = 0.f;
    #pragma unroll 1
    for (int k = 0; k < DPAD; ++k) norm += f[k]*f[k];
    if (side) ny[(size_t)b*NPTS + n] = norm;
    else      nx[(size_t)b*NPTS + n] = norm;

    int tile = b*TILES + (n >> 4);   // global tile index
    int m    = n & 15;               // row/col within tile

    #pragma unroll 1
    for (int K = 0; K < DPAD; ++K) {
        int kb = K >> 5, ko = K & 31;
        float  v = f[K];
        bf16_t h = (bf16_t)v;
        bf16_t l = (bf16_t)(v - (float)h);
        int lane, idx;
        if (side == 0) {
            // A-matrix 16x32 bf16 layout (ISA 7.12.2):
            // lanes 0-15 : elem 0-7 = K0..7,  elem 8-15 = K16..23
            // lanes 16-31: elem 0-7 = K8..15, elem 8-15 = K24..31
            lane = m + ((ko & 8)  ? 16 : 0);
            idx  = (ko & 7) + ((ko & 16) ? 8 : 0);
        } else {
            // B-matrix 32x16 bf16 layout (same K order as A):
            // lanes 0-15 : elem 0-7 = K0..7,  elem 8-15 = K16..23
            // lanes 16-31: elem 0-7 = K8..15, elem 8-15 = K24..31
            lane = m + ((ko & 8)  ? 16 : 0);
            idx  = (ko & 7) + ((ko & 16) ? 8 : 0);
        }
        size_t addr = (((size_t)tile*KB + kb)*32 + lane)*16 + idx;
        if (side == 0) { Axh[addr] = h; Axl[addr] = l; }
        else           { Byh[addr] = h; Byl[addr] = l; }
    }
}

// ---------------------------------------------------------------------------
// Kernel 2: zero the scalar accumulator (d_out is poisoned by the harness).
// ---------------------------------------------------------------------------
__global__ void zero_sum(float* out)
{
    if (threadIdx.x == 0 && blockIdx.x == 0) out[0] = 0.f;
}

// ---------------------------------------------------------------------------
// RBF epilogue for one 16x16 accumulator tile.
// ---------------------------------------------------------------------------
__device__ __forceinline__ float epilogue(v8f acc, int b, int rowbase, int col,
                                          float nyv,
                                          const float* __restrict__ nx,
                                          float* __restrict__ gram)
{
    float psum = 0.f;
    #pragma unroll
    for (int r = 0; r < 8; ++r) {
        int row  = rowbase + r;
        float sq = nx[(size_t)b*NPTS + row] + nyv - 2.0f*acc[r];
        sq = fmaxf(sq, 0.0f);
        float g = __expf(-sq * INV_2ELL2);
        g = (g > THRESH) ? g : 0.0f;
        gram[(((size_t)b*NPTS + row) * NPTS) + col] = g;
        psum += g;
    }
    return psum;
}

// ---------------------------------------------------------------------------
// Kernel 3: one wave per 32x32 gram patch (2 M-tiles x 2 N-tiles) to halve
// packed-operand load traffic via register reuse. bf16x3 WMMA over K=128,
// fused RBF epilogue, LDS block reduction, one global atomic per block.
// grid = (128, 16, BATCH), block = 256 threads = 8 waves.
// ---------------------------------------------------------------------------
__global__ void __launch_bounds__(256)
gram_wmma(const bf16_t* __restrict__ Axh, const bf16_t* __restrict__ Axl,
          const bf16_t* __restrict__ Byh, const bf16_t* __restrict__ Byl,
          const float* __restrict__ nx,  const float* __restrict__ ny,
          float* __restrict__ out)
{
    __shared__ float bsum;
    if (threadIdx.x == 0) bsum = 0.f;
    __syncthreads();

    int lane = threadIdx.x & 31;
    int wave = threadIdx.x >> 5;
    int mt0  = blockIdx.x * 2;                 // first of 2 M-tiles
    int nt0  = (blockIdx.y * 8 + wave) * 2;    // first of 2 N-tiles
    int b    = blockIdx.z;

    const v16bf* pAh = (const v16bf*)Axh + ((size_t)(b*TILES + mt0)*KB)*32 + lane;
    const v16bf* pAl = (const v16bf*)Axl + ((size_t)(b*TILES + mt0)*KB)*32 + lane;
    const v16bf* pBh = (const v16bf*)Byh + ((size_t)(b*TILES + nt0)*KB)*32 + lane;
    const v16bf* pBl = (const v16bf*)Byl + ((size_t)(b*TILES + nt0)*KB)*32 + lane;

    v8f acc00 = {}, acc01 = {}, acc10 = {}, acc11 = {};

    #pragma unroll
    for (int kb = 0; kb < KB; ++kb) {
        const int o0 = kb*32;            // tile +0
        const int o1 = (KB + kb)*32;     // tile +1

        v16bf a0h = pAh[o0], a0l = pAl[o0];
        v16bf a1h = pAh[o1], a1l = pAl[o1];
        v16bf b0h = pBh[o0], b0l = pBl[o0];
        v16bf b1h = pBh[o1], b1l = pBl[o1];

        acc00 = __builtin_amdgcn_wmma_f32_16x16x32_bf16(false, a0h, false, b0h, (short)0, acc00, false, false);
        acc00 = __builtin_amdgcn_wmma_f32_16x16x32_bf16(false, a0h, false, b0l, (short)0, acc00, false, false);
        acc00 = __builtin_amdgcn_wmma_f32_16x16x32_bf16(false, a0l, false, b0h, (short)0, acc00, false, false);

        acc01 = __builtin_amdgcn_wmma_f32_16x16x32_bf16(false, a0h, false, b1h, (short)0, acc01, false, false);
        acc01 = __builtin_amdgcn_wmma_f32_16x16x32_bf16(false, a0h, false, b1l, (short)0, acc01, false, false);
        acc01 = __builtin_amdgcn_wmma_f32_16x16x32_bf16(false, a0l, false, b1h, (short)0, acc01, false, false);

        acc10 = __builtin_amdgcn_wmma_f32_16x16x32_bf16(false, a1h, false, b0h, (short)0, acc10, false, false);
        acc10 = __builtin_amdgcn_wmma_f32_16x16x32_bf16(false, a1h, false, b0l, (short)0, acc10, false, false);
        acc10 = __builtin_amdgcn_wmma_f32_16x16x32_bf16(false, a1l, false, b0h, (short)0, acc10, false, false);

        acc11 = __builtin_amdgcn_wmma_f32_16x16x32_bf16(false, a1h, false, b1h, (short)0, acc11, false, false);
        acc11 = __builtin_amdgcn_wmma_f32_16x16x32_bf16(false, a1h, false, b1l, (short)0, acc11, false, false);
        acc11 = __builtin_amdgcn_wmma_f32_16x16x32_bf16(false, a1l, false, b1h, (short)0, acc11, false, false);
    }

    // C/D 16x16 f32 layout: VGPR r, lanes 0-15 -> (M=r, N=lane);
    //                       lanes 16-31 -> (M=r+8, N=lane-16)
    int col0 = nt0*16 + (lane & 15);
    int col1 = col0 + 16;
    int rb0  = mt0*16 + ((lane >> 4) << 3);
    int rb1  = rb0 + 16;
    float ny0 = ny[(size_t)b*NPTS + col0];
    float ny1 = ny[(size_t)b*NPTS + col1];
    float* gram = out + 1;

    float psum = 0.f;
    psum += epilogue(acc00, b, rb0, col0, ny0, nx, gram);
    psum += epilogue(acc01, b, rb0, col1, ny1, nx, gram);
    psum += epilogue(acc10, b, rb1, col0, ny0, nx, gram);
    psum += epilogue(acc11, b, rb1, col1, ny1, nx, gram);

    #pragma unroll
    for (int off = 16; off > 0; off >>= 1)
        psum += __shfl_xor(psum, off, 32);

    if (lane == 0) atomicAdd(&bsum, psum);   // ds_add_f32, 8 per block
    __syncthreads();
    if (threadIdx.x == 0) atomicAdd(out, bsum);  // 1 global atomic per block
}

// ---------------------------------------------------------------------------
extern "C" void kernel_launch(void* const* d_in, const int* in_sizes, int n_in,
                              void* d_out, int out_size, void* d_ws, size_t ws_size,
                              hipStream_t stream) {
    const float* pc1   = (const float*)d_in[0];
    const float* pc2   = (const float*)d_in[1];
    const float* xfeat = (const float*)d_in[2];
    const float* yfeat = (const float*)d_in[3];
    const float* Tq    = (const float*)d_in[4];
    float* out = (float*)d_out;

    bf16_t* Axh = (bf16_t*)d_ws;
    bf16_t* Axl = Axh + PACK_ELEMS;
    bf16_t* Byh = Axl + PACK_ELEMS;
    bf16_t* Byl = Byh + PACK_ELEMS;
    float*  nx  = (float*)(Byl + PACK_ELEMS);
    float*  ny  = nx + (size_t)BATCH * NPTS;

    prep_pack<<<64, 256, 0, stream>>>(pc1, pc2, xfeat, yfeat, Tq,
                                      Axh, Axl, Byh, Byl, nx, ny);
    zero_sum<<<1, 32, 0, stream>>>(out);
    dim3 grid(TILES/2, TILES/2/8, BATCH);
    gram_wmma<<<grid, 256, 0, stream>>>(Axh, Axl, Byh, Byl, nx, ny, out);
}
